// AttentionLayer_34746285425235
// MI455X (gfx1250) — compile-verified, matrix operand order checked by reading, not run on hardware
//
#include <hip/hip_runtime.h>

// AttentionLayer for MI455X (gfx1250, wave32).
// All four GEMMs use v_wmma_f32_16x16x32_f16 (f16 in, f32 accumulate).
// Attention is flash-style: S is computed TRANSPOSED (S^T = K.Q^T) so that
//  - softmax row-stats are per-lane (one shfl_xor(16) each),
//  - exp(S^T) lands directly in the WMMA B-fragment register layout (no
//    LDS/shuffle transpose in the inner loop),
//  - V is stored transposed so the P.V A-fragments load contiguously.

#define BB   2
#define LL   2048
#define HHS  1024
#define NHH  16
#define HDD  64
#define WPB  8   // waves per block in attention kernel

typedef __attribute__((ext_vector_type(16))) _Float16 v16h;
typedef __attribute__((ext_vector_type(8)))  _Float16 v8h;
typedef __attribute__((ext_vector_type(8)))  float    v8f;

__device__ __forceinline__ v8f wmma16(v16h a, v16h b, v8f c) {
  // (neg_a, A, neg_b, B, c_mod, C, reuse_a, reuse_b)
  return __builtin_amdgcn_wmma_f32_16x16x32_f16(false, a, false, b, (short)0, c,
                                                false, false);
}

// Load a 16x32 f16 A-fragment (or a B-fragment for C = A.W^T, where the
// B-matrix column n is row n of W, contiguous along K).
// Per the CDNA5 16-bit A layout: lane holds row (lane&15); VGPRs 0-3 carry
// K = half*8 + 0..7 and VGPRs 4-7 carry K = 16 + half*8 + 0..7 (half = lane>>4),
// i.e. two contiguous 16-byte chunks per lane.
__device__ __forceinline__ v16h load_frag(const _Float16* __restrict__ base,
                                          int row0, int ld, int k0) {
  const int lane = threadIdx.x & 31;
  const _Float16* p =
      base + (size_t)(row0 + (lane & 15)) * ld + k0 + ((lane >> 4) << 3);
  v8h lo = *(const v8h*)(p);
  v8h hi = *(const v8h*)(p + 16);
  v16h r;
#pragma unroll
  for (int i = 0; i < 8; ++i) { r[i] = lo[i]; r[8 + i] = hi[i]; }
  return r;
}

// One wave computes a 64x32 tile of C = A(MxK) . W(NxK)^T, f32 accum.
__device__ __forceinline__ void gemm64x32(const _Float16* __restrict__ A, int lda,
                                          const _Float16* __restrict__ W, int ldw,
                                          int m0, int n0, int ktot, v8f c[4][2]) {
  for (int k = 0; k < ktot; k += 32) {
    v16h b0 = load_frag(W, n0,      ldw, k);
    v16h b1 = load_frag(W, n0 + 16, ldw, k);
#pragma unroll
    for (int mi = 0; mi < 4; ++mi) {
      v16h a = load_frag(A, m0 + mi * 16, lda, k);
      c[mi][0] = wmma16(a, b0, c[mi][0]);
      c[mi][1] = wmma16(a, b1, c[mi][1]);
    }
  }
}

__global__ void cvt_f32_f16(const float* __restrict__ in,
                            _Float16* __restrict__ out, int n) {
  int i = (blockIdx.x * blockDim.x + threadIdx.x) * 4;
  if (i + 3 < n) {
    float4 f = *(const float4*)(in + i);
    out[i]     = (_Float16)f.x;
    out[i + 1] = (_Float16)f.y;
    out[i + 2] = (_Float16)f.z;
    out[i + 3] = (_Float16)f.w;
  } else {
    for (; i < n; ++i) out[i] = (_Float16)in[i];
  }
}

// qkv = x @ w_qkv^T + b_qkv, scattered to Q (pre-scaled by HD^-0.5),
// K [b,h,l,d] and V transposed [b,h,d,l].
__global__ __launch_bounds__(32) void qkv_gemm_kernel(
    const _Float16* __restrict__ X, const _Float16* __restrict__ W,
    const float* __restrict__ bias, _Float16* __restrict__ Qo,
    _Float16* __restrict__ Ko, _Float16* __restrict__ Vto) {
  const int n0 = blockIdx.x * 32;
  const int m0 = blockIdx.y * 64;
  const v8f vz = {0.f, 0.f, 0.f, 0.f, 0.f, 0.f, 0.f, 0.f};
  v8f c[4][2];
#pragma unroll
  for (int mi = 0; mi < 4; ++mi) { c[mi][0] = vz; c[mi][1] = vz; }

  gemm64x32(X, HHS, W, HHS, m0, n0, HHS, c);

  const int lane = threadIdx.x & 31;
  const int half = lane >> 4, nlo = lane & 15;
#pragma unroll
  for (int ni = 0; ni < 2; ++ni) {
    const int n = n0 + ni * 16 + nlo;
    const float bv = bias[n];
    const int region = n >> 10;          // 0:Q 1:K 2:V
    const int h = (n >> 6) & (NHH - 1);
    const int d = n & (HDD - 1);
#pragma unroll
    for (int mi = 0; mi < 4; ++mi) {
#pragma unroll
      for (int v = 0; v < 8; ++v) {
        const int m = m0 + mi * 16 + half * 8 + v;
        const int b = m >> 11;           // L = 2048
        const int lpos = m & (LL - 1);
        const float val = c[mi][ni][v] + bv;
        const size_t bhl = (size_t)(b * NHH + h) * LL + lpos;
        if (region == 0)
          Qo[bhl * HDD + d] = (_Float16)(val * 0.125f);   // fold softmax scale
        else if (region == 1)
          Ko[bhl * HDD + d] = (_Float16)val;
        else
          Vto[((size_t)(b * NHH + h) * HDD + d) * LL + lpos] = (_Float16)val;
      }
    }
  }
}

// Flash attention, one wave per 16-query tile. Computes S^T = K.Q^T so the
// probability tile is already in B-fragment layout for O^T += Vt . P^T.
__global__ __launch_bounds__(WPB * 32) void attn_kernel(
    const _Float16* __restrict__ Q, const _Float16* __restrict__ K,
    const _Float16* __restrict__ Vt, _Float16* __restrict__ Ctx) {
  const int lane = threadIdx.x & 31;
  const int wave = threadIdx.x >> 5;
  const int qt = blockIdx.x * WPB + wave;
  const int tiles = LL / 16;
  const int bh = qt / tiles;
  const int q0 = (qt % tiles) * 16;

  const _Float16* Qb = Q  + (size_t)bh * LL * HDD;
  const _Float16* Kb = K  + (size_t)bh * LL * HDD;
  const _Float16* Vb = Vt + (size_t)bh * HDD * LL;

  // Q as B-fragments (column q = row q of Q, contiguous along d).
  const v16h qf0 = load_frag(Qb, q0, HDD, 0);
  const v16h qf1 = load_frag(Qb, q0, HDD, 32);

  const v8f vz = {0.f, 0.f, 0.f, 0.f, 0.f, 0.f, 0.f, 0.f};
  v8f o[4];
#pragma unroll
  for (int i = 0; i < 4; ++i) o[i] = vz;
  float m_run = -1e30f, l_run = 0.f;

  for (int k0 = 0; k0 < LL; k0 += 32) {
    // S^T tiles: keys k0..k0+15 (s0) and k0+16..k0+31 (s1); columns = queries.
    v8f s0 = vz, s1 = vz;
    v16h ka;
    ka = load_frag(Kb, k0,      HDD, 0);  s0 = wmma16(ka, qf0, s0);
    ka = load_frag(Kb, k0,      HDD, 32); s0 = wmma16(ka, qf1, s0);
    ka = load_frag(Kb, k0 + 16, HDD, 0);  s1 = wmma16(ka, qf0, s1);
    ka = load_frag(Kb, k0 + 16, HDD, 32); s1 = wmma16(ka, qf1, s1);

    // Online softmax: each lane owns column q = lane&15; lanes q and q+16
    // hold complementary key halves -> one shfl_xor(16) combines them.
    float mx = m_run;
#pragma unroll
    for (int v = 0; v < 8; ++v) mx = fmaxf(mx, fmaxf(s0[v], s1[v]));
    mx = fmaxf(mx, __shfl_xor(mx, 16));

    const float corr = __expf(m_run - mx);
    float rs = 0.f;
    v16h pb;  // P^T directly in WMMA B-fragment element order
#pragma unroll
    for (int v = 0; v < 8; ++v) {
      const float p0 = __expf(s0[v] - mx);
      const float p1 = __expf(s1[v] - mx);
      rs += p0 + p1;
      pb[v]     = (_Float16)p0;
      pb[8 + v] = (_Float16)p1;
    }
    rs += __shfl_xor(rs, 16);
    l_run = l_run * corr + rs;
    m_run = mx;

#pragma unroll
    for (int di = 0; di < 4; ++di)
#pragma unroll
      for (int j = 0; j < 8; ++j) o[di][j] *= corr;   // per-lane uniform scale

    // O^T += Vt(16d x 32k) . P^T(32k x 16q)
#pragma unroll
    for (int di = 0; di < 4; ++di) {
      v16h va = load_frag(Vb, di * 16, LL, k0);
      o[di] = wmma16(va, pb, o[di]);
    }
  }

  const float inv_l = 1.f / l_run;
  const int qrow = q0 + (lane & 15);
  const int b = bh / NHH, h = bh % NHH;
  const int half = lane >> 4;
  _Float16* cp = Ctx + ((size_t)b * LL + qrow) * HHS + h * HDD;
#pragma unroll
  for (int di = 0; di < 4; ++di)
#pragma unroll
    for (int v = 0; v < 8; ++v)
      cp[di * 16 + half * 8 + v] = (_Float16)(o[di][v] * inv_l);
}

// out = ctx @ w_out^T + b_out (fp32 output)
__global__ __launch_bounds__(32) void out_gemm_kernel(
    const _Float16* __restrict__ Ctx, const _Float16* __restrict__ Wo,
    const float* __restrict__ bias, float* __restrict__ out) {
  const int n0 = blockIdx.x * 32;
  const int m0 = blockIdx.y * 64;
  const v8f vz = {0.f, 0.f, 0.f, 0.f, 0.f, 0.f, 0.f, 0.f};
  v8f c[4][2];
#pragma unroll
  for (int mi = 0; mi < 4; ++mi) { c[mi][0] = vz; c[mi][1] = vz; }

  gemm64x32(Ctx, HHS, Wo, HHS, m0, n0, HHS, c);

  const int lane = threadIdx.x & 31;
  const int half = lane >> 4, nlo = lane & 15;
#pragma unroll
  for (int ni = 0; ni < 2; ++ni) {
    const int n = n0 + ni * 16 + nlo;
    const float bv = bias[n];
#pragma unroll
    for (int mi = 0; mi < 4; ++mi)
#pragma unroll
      for (int v = 0; v < 8; ++v) {
        const int m = m0 + mi * 16 + half * 8 + v;
        out[(size_t)m * HHS + n] = c[mi][ni][v] + bv;
      }
  }
}

extern "C" void kernel_launch(void* const* d_in, const int* in_sizes, int n_in,
                              void* d_out, int out_size, void* d_ws, size_t ws_size,
                              hipStream_t stream) {
  const float* x     = (const float*)d_in[0];
  const float* w_qkv = (const float*)d_in[1];
  const float* b_qkv = (const float*)d_in[2];
  const float* w_out = (const float*)d_in[3];
  const float* b_out = (const float*)d_in[4];
  float* out = (float*)d_out;

  const size_t nX  = (size_t)BB * LL * HHS;  // 4,194,304
  const size_t nWq = (size_t)3 * HHS * HHS;  // 3,145,728
  const size_t nWo = (size_t)HHS * HHS;      // 1,048,576

  _Float16* p = (_Float16*)d_ws;             // ~48 MB f16 total
  _Float16* Xh   = p; p += nX;
  _Float16* Wqh  = p; p += nWq;
  _Float16* Woh  = p; p += nWo;
  _Float16* Qh   = p; p += nX;
  _Float16* Kh   = p; p += nX;
  _Float16* Vth  = p; p += nX;
  _Float16* Ctxh = p; p += nX;

  {
    int blocks = (int)((nX / 4 + 255) / 256);
    cvt_f32_f16<<<blocks, 256, 0, stream>>>(x, Xh, (int)nX);
    blocks = (int)((nWq / 4 + 255) / 256);
    cvt_f32_f16<<<blocks, 256, 0, stream>>>(w_qkv, Wqh, (int)nWq);
    blocks = (int)((nWo / 4 + 255) / 256);
    cvt_f32_f16<<<blocks, 256, 0, stream>>>(w_out, Woh, (int)nWo);
  }

  qkv_gemm_kernel<<<dim3(3 * HHS / 32, BB * LL / 64), 32, 0, stream>>>(
      Xh, Wqh, b_qkv, Qh, Kh, Vth);

  attn_kernel<<<(BB * NHH * (LL / 16)) / WPB, WPB * 32, 0, stream>>>(
      Qh, Kh, Vth, Ctxh);

  out_gemm_kernel<<<dim3(HHS / 32, BB * LL / 64), 32, 0, stream>>>(
      Ctxh, Woh, b_out, out);
}
